// MultiHeadAttention_69226282876940
// MI455X (gfx1250) — matrix operand order repaired and checked by reading, not compiled, not run on hardware
//
#include <hip/hip_runtime.h>
#include <hip/hip_bf16.h>

// ---------------------------------------------------------------------------
// MHA block for gfx1250 (MI455X).
//  - all GEMMs: v_wmma_f32_16x16x32_bf16 (fp32 accum)
//  - activations/weights pre-converted to bf16 (weights transposed) so every
//    WMMA fragment load is contiguous bf16 -> global_load_b128, no cvt in
//    the inner loops
//  - attention: 32x2048 score panel resident in 256KB LDS; mask applied in a
//    coalesced dword pass; softmax in LDS; 512MB attn tensor written exactly
//    once with non-temporal stores
// ---------------------------------------------------------------------------

typedef __attribute__((ext_vector_type(16))) __bf16 v16bf;
typedef __attribute__((ext_vector_type(8)))  float  v8f;

#define NHEAD 8
#define DMODEL 512
#define DHEAD 64
#define BATCH 4
#define SEQ 2048
#define ROWS (BATCH * SEQ)          // 8192

// ---------------------------------------------------------------------------
// K0a: elementwise f32 -> bf16 (activations), 8 elems/thread
// ---------------------------------------------------------------------------
__global__ void convert_bf16(const float* __restrict__ in,
                             __bf16* __restrict__ out, int n) {
    int base = (blockIdx.x * blockDim.x + threadIdx.x) * 8;
    if (base + 8 <= n) {
#pragma unroll
        for (int i = 0; i < 8; ++i) out[base + i] = (__bf16)in[base + i];
    }
}

// ---------------------------------------------------------------------------
// K0b: weight convert + transpose: Wt[n*512+k] = bf16(W[k*512+n])
// ---------------------------------------------------------------------------
__global__ void transpose_w_bf16(const float* __restrict__ W,
                                 __bf16* __restrict__ Wt) {
    int idx = blockIdx.x * blockDim.x + threadIdx.x;   // 0 .. 512*512-1
    int n = idx >> 9;
    int k = idx & 511;
    Wt[(size_t)n * DMODEL + k] = (__bf16)W[(size_t)k * DMODEL + n];
}

// ---------------------------------------------------------------------------
// K1: C = A_bf16[8192,512] @ W  (Wt stored [n][k] bf16)
// one wave -> 16x64 output strip, A frag reused across 4 WMMAs per K step.
// VMODE==0: C row-major bf16 [8192,512]
// VMODE==1: V-transposed layout Vt[((b*8+h)*64+d)*2048 + kpos]
// ---------------------------------------------------------------------------
template <int VMODE>
__global__ void proj_gemm_bf16(const __bf16* __restrict__ A,
                               const __bf16* __restrict__ Wt,
                               __bf16* __restrict__ Cb) {
    const int lane = threadIdx.x & 31;
    const int wv   = threadIdx.x >> 5;
    const int g    = blockIdx.x * 8 + wv;
    const int mtile = g >> 3;       // 0..511  (M/16)
    const int n64   = g & 7;        // 0..7    (N/64)
    const int half = lane >> 4;
    const int lm   = lane & 15;

    v8f acc[4] = {v8f{}, v8f{}, v8f{}, v8f{}};
    const __bf16* Arow = A + (size_t)(mtile * 16 + lm) * DMODEL;
#pragma unroll 2
    for (int kk = 0; kk < DMODEL; kk += 32) {
        v16bf a;
#pragma unroll
        for (int e = 0; e < 16; ++e) {
            int K = kk + half * 16 + e;
            a[e] = Arow[K];
        }
#pragma unroll
        for (int j = 0; j < 4; ++j) {
            const __bf16* Wrow = Wt + (size_t)(n64 * 64 + j * 16 + lm) * DMODEL;
            v16bf bb;
#pragma unroll
            for (int e = 0; e < 16; ++e) bb[e] = Wrow[kk + half * 16 + e];
            acc[j] = __builtin_amdgcn_wmma_f32_16x16x32_bf16(
                false, a, false, bb, (short)0, acc[j], false, false);
        }
    }
#pragma unroll
    for (int j = 0; j < 4; ++j) {
        int n = n64 * 64 + j * 16 + lm;
#pragma unroll
        for (int r = 0; r < 8; ++r) {
            int m = mtile * 16 + half * 8 + r;
            __bf16 val = (__bf16)acc[j][r];
            if (VMODE == 0) {
                Cb[(size_t)m * DMODEL + n] = val;
            } else {
                // Vt[((b*8+h)*64 + d)*2048 + kpos]
                int b = m >> 11, kpos = m & 2047;
                int h = n >> 6,  d    = n & 63;
                Cb[(((size_t)(b * NHEAD + h) * DHEAD + d) << 11) + kpos] = val;
            }
        }
    }
}

// ---------------------------------------------------------------------------
// K2: fused attention.  One workgroup per (b, h, 32-row query block).
// LDS: S[32][2048] f32 (256KB) + reduction scratch.
//   phase1 : S = QK^T / 8                       (WMMA)
//   phase1b: coalesced dword mask pass          (contiguous 64KB mask block)
//   phase2 : softmax rows in LDS, non-temporal stream of normalized attn
//   phase3 : O = P @ V                          (WMMA, P from LDS)
// ---------------------------------------------------------------------------
__global__ void attn_fused(const __bf16* __restrict__ Qh,
                           const __bf16* __restrict__ Kh,
                           const __bf16* __restrict__ Vt,
                           const unsigned char* __restrict__ mask,
                           float* __restrict__ attn_out,
                           __bf16* __restrict__ Ob) {
    extern __shared__ float smem[];
    float* S       = smem;                  // 32*2048
    float* partial = smem + 32 * SEQ;       // 256
    float* rmax    = partial + 256;         // 32
    float* rinv    = rmax + 32;             // 32

    const int qb = blockIdx.x & 63;         // query block (32 rows each)
    const int h  = (blockIdx.x >> 6) & 7;
    const int b  = blockIdx.x >> 9;
    const int t    = threadIdx.x;
    const int lane = t & 31;
    const int wv   = t >> 5;
    const int half = lane >> 4;
    const int lm   = lane & 15;

    // ---- phase 1: scores S = QK^T / 8 ----
    const int mt = wv & 1;
    const __bf16* Qrow =
        Qh + (size_t)(b * SEQ + qb * 32 + mt * 16 + lm) * DMODEL + h * DHEAD;
    v16bf a0, a1;                           // Q fragment, d = 0..63
#pragma unroll
    for (int e = 0; e < 16; ++e) {
        int d = half * 16 + e;
        a0[e] = Qrow[d];
        a1[e] = Qrow[32 + d];
    }
    for (int nt = (wv >> 1); nt < SEQ / 16; nt += 4) {
        const int n0 = nt * 16;
        const __bf16* Krow =
            Kh + (size_t)(b * SEQ + n0 + lm) * DMODEL + h * DHEAD;
        v16bf b0, b1;
#pragma unroll
        for (int e = 0; e < 16; ++e) {
            int d = half * 16 + e;
            b0[e] = Krow[d];
            b1[e] = Krow[32 + d];
        }
        v8f acc = {};
        acc = __builtin_amdgcn_wmma_f32_16x16x32_bf16(false, a0, false, b0,
                                                      (short)0, acc, false, false);
        acc = __builtin_amdgcn_wmma_f32_16x16x32_bf16(false, a1, false, b1,
                                                      (short)0, acc, false, false);
#pragma unroll
        for (int r = 0; r < 8; ++r) {
            int ml = mt * 16 + half * 8 + r;
            S[ml * SEQ + n0 + lm] = acc[r] * 0.125f;   // 1/sqrt(64)
        }
    }
    __syncthreads();

    // ---- phase 1b: mask (contiguous 64KB block, coalesced dword loads) ----
    {
        const unsigned* m32 = (const unsigned*)(mask +
            (size_t)(b * SEQ + qb * 32) * SEQ);
        for (int i = t; i < (32 * SEQ) / 4; i += 256) {
            unsigned w = m32[i];
            if (w != 0u) {
#pragma unroll
                for (int j = 0; j < 4; ++j)
                    if ((w >> (8 * j)) & 0xffu) S[i * 4 + j] = -1e10f;
            }
        }
    }
    __syncthreads();

    // ---- phase 2: softmax rows in LDS ----
    {
        const int row = t >> 3, sub = t & 7;
        float mx = -3.4e38f;
        for (int c = sub; c < SEQ; c += 8) mx = fmaxf(mx, S[row * SEQ + c]);
        partial[t] = mx;
        __syncthreads();
        if (sub == 0) {
            float m2 = partial[t];
#pragma unroll
            for (int i = 1; i < 8; ++i) m2 = fmaxf(m2, partial[t + i]);
            rmax[row] = m2;
        }
        __syncthreads();
        const float rm = rmax[row];
        float sum = 0.f;
        for (int c = sub; c < SEQ; c += 8) {
            float e = __expf(S[row * SEQ + c] - rm);
            S[row * SEQ + c] = e;
            sum += e;
        }
        partial[t] = sum;
        __syncthreads();
        if (sub == 0) {
            float s2 = 0.f;
#pragma unroll
            for (int i = 0; i < 8; ++i) s2 += partial[t + i];
            rinv[row] = 1.0f / s2;
        }
        __syncthreads();
        // normalize in LDS + coalesced non-temporal stream of attn
        float* dst = attn_out +
            (((size_t)h * BATCH + b) * SEQ + (size_t)qb * 32) * SEQ;
        for (int idx = t; idx < 32 * SEQ; idx += 256) {
            int r2 = idx >> 11;
            float v = S[idx] * rinv[r2];
            S[idx] = v;
            __builtin_nontemporal_store(v, dst + idx);
        }
    }
    __syncthreads();

    // ---- phase 3: O = P @ V  (8 tiles: 2 mtiles x 4 d-tiles, one per wave) ----
    {
        const int mt3 = wv >> 2;
        const int dn  = wv & 3;
        const __bf16* Vrow =
            Vt + (((size_t)(b * NHEAD + h) * DHEAD + dn * 16 + lm) << 11);
        v8f acc = {};
#pragma unroll 2
        for (int kp = 0; kp < SEQ; kp += 32) {
            v16bf a;
#pragma unroll
            for (int e = 0; e < 16; ++e) {
                int K = kp + half * 16 + e;
                a[e] = (__bf16)S[(mt3 * 16 + lm) * SEQ + K];
            }
            v16bf bb;
#pragma unroll
            for (int e = 0; e < 16; ++e) bb[e] = Vrow[kp + half * 16 + e];
            acc = __builtin_amdgcn_wmma_f32_16x16x32_bf16(
                false, a, false, bb, (short)0, acc, false, false);
        }
#pragma unroll
        for (int r = 0; r < 8; ++r) {
            int rowg = b * SEQ + qb * 32 + mt3 * 16 + half * 8 + r;
            int col  = h * DHEAD + dn * 16 + lm;
            Ob[(size_t)rowg * DMODEL + col] = (__bf16)acc[r];
        }
    }
}

// ---------------------------------------------------------------------------
// K3: X = O_bf16 @ Wfc + bias + residual   (f32 into d_out y-region)
// ---------------------------------------------------------------------------
__global__ void fc_residual(const __bf16* __restrict__ Ob,
                            const __bf16* __restrict__ WtFc,
                            const float* __restrict__ bias,
                            const float* __restrict__ resid,
                            float* __restrict__ X) {
    const int lane = threadIdx.x & 31;
    const int wv   = threadIdx.x >> 5;
    const int g    = blockIdx.x * 8 + wv;
    const int mtile = g >> 3;
    const int n64   = g & 7;
    const int half = lane >> 4;
    const int lm   = lane & 15;

    v8f acc[4] = {v8f{}, v8f{}, v8f{}, v8f{}};
    const __bf16* Arow = Ob + (size_t)(mtile * 16 + lm) * DMODEL;
#pragma unroll 2
    for (int kk = 0; kk < DMODEL; kk += 32) {
        v16bf a;
#pragma unroll
        for (int e = 0; e < 16; ++e) {
            int K = kk + half * 16 + e;
            a[e] = Arow[K];
        }
#pragma unroll
        for (int j = 0; j < 4; ++j) {
            const __bf16* Wrow = WtFc + (size_t)(n64 * 64 + j * 16 + lm) * DMODEL;
            v16bf bb;
#pragma unroll
            for (int e = 0; e < 16; ++e) bb[e] = Wrow[kk + half * 16 + e];
            acc[j] = __builtin_amdgcn_wmma_f32_16x16x32_bf16(
                false, a, false, bb, (short)0, acc[j], false, false);
        }
    }
#pragma unroll
    for (int j = 0; j < 4; ++j) {
        int n = n64 * 64 + j * 16 + lm;
#pragma unroll
        for (int r = 0; r < 8; ++r) {
            int m = mtile * 16 + half * 8 + r;
            X[(size_t)m * DMODEL + n] =
                acc[j][r] + bias[n] + resid[(size_t)m * DMODEL + n];
        }
    }
}

// ---------------------------------------------------------------------------
// K4: in-place LayerNorm over each 512-wide row of d_out's y region
// ---------------------------------------------------------------------------
__global__ void layernorm_k(float* __restrict__ Y,
                            const float* __restrict__ gm,
                            const float* __restrict__ bt) {
    __shared__ float red[256];
    __shared__ float red2[256];
    const int row = blockIdx.x;
    const int t = threadIdx.x;
    float x0 = Y[(size_t)row * DMODEL + t];
    float x1 = Y[(size_t)row * DMODEL + t + 256];
    red[t]  = x0 + x1;
    red2[t] = x0 * x0 + x1 * x1;
    __syncthreads();
    for (int s = 128; s > 0; s >>= 1) {
        if (t < s) { red[t] += red[t + s]; red2[t] += red2[t + s]; }
        __syncthreads();
    }
    float mu  = red[0] * (1.0f / DMODEL);
    float var = red2[0] * (1.0f / DMODEL) - mu * mu;
    float rs  = rsqrtf(var + 1e-5f);
    Y[(size_t)row * DMODEL + t]       = (x0 - mu) * rs * gm[t] + bt[t];
    Y[(size_t)row * DMODEL + t + 256] = (x1 - mu) * rs * gm[t + 256] + bt[t + 256];
}

// ---------------------------------------------------------------------------
extern "C" void kernel_launch(void* const* d_in, const int* in_sizes, int n_in,
                              void* d_out, int out_size, void* d_ws, size_t ws_size,
                              hipStream_t stream) {
    const float* q     = (const float*)d_in[0];
    const float* k     = (const float*)d_in[1];
    const float* v     = (const float*)d_in[2];
    const unsigned char* mask = (const unsigned char*)d_in[3];
    const float* w_q   = (const float*)d_in[4];
    const float* w_k   = (const float*)d_in[5];
    const float* w_v   = (const float*)d_in[6];
    const float* w_fc  = (const float*)d_in[7];
    const float* b_fc  = (const float*)d_in[8];
    const float* gamma = (const float*)d_in[9];
    const float* beta  = (const float*)d_in[10];

    const size_t ACT = (size_t)ROWS * DMODEL;             // 4,194,304 elems
    __bf16* Xq  = (__bf16*)d_ws;
    __bf16* Xk  = Xq + ACT;
    __bf16* Xv  = Xk + ACT;
    __bf16* Qh  = Xv + ACT;
    __bf16* Kh  = Qh + ACT;
    __bf16* Vt  = Kh + ACT;
    __bf16* Ob  = Vt + ACT;
    __bf16* Wqt = Ob + ACT;
    __bf16* Wkt = Wqt + (size_t)DMODEL * DMODEL;
    __bf16* Wvt = Wkt + (size_t)DMODEL * DMODEL;
    __bf16* Wft = Wvt + (size_t)DMODEL * DMODEL;

    float* y    = (float*)d_out;                          // [4,2048,512]
    float* attn = y + (size_t)BATCH * SEQ * DMODEL;       // [32,2048,2048]

    // pre-convert activations (bf16) and weights (bf16, transposed)
    convert_bf16<<<(int)(ACT / (256 * 8)), 256, 0, stream>>>(q, Xq, (int)ACT);
    convert_bf16<<<(int)(ACT / (256 * 8)), 256, 0, stream>>>(k, Xk, (int)ACT);
    convert_bf16<<<(int)(ACT / (256 * 8)), 256, 0, stream>>>(v, Xv, (int)ACT);
    transpose_w_bf16<<<(DMODEL * DMODEL) / 256, 256, 0, stream>>>(w_q, Wqt);
    transpose_w_bf16<<<(DMODEL * DMODEL) / 256, 256, 0, stream>>>(w_k, Wkt);
    transpose_w_bf16<<<(DMODEL * DMODEL) / 256, 256, 0, stream>>>(w_v, Wvt);
    transpose_w_bf16<<<(DMODEL * DMODEL) / 256, 256, 0, stream>>>(w_fc, Wft);

    // QKV projections (V written transposed for contiguous P@V fragments)
    proj_gemm_bf16<0><<<512, 256, 0, stream>>>(Xq, Wqt, Qh);
    proj_gemm_bf16<0><<<512, 256, 0, stream>>>(Xk, Wkt, Kh);
    proj_gemm_bf16<1><<<512, 256, 0, stream>>>(Xv, Wvt, Vt);

    // fused scores + softmax + attn-write + P@V
    size_t smem = (size_t)(32 * SEQ + 256 + 32 + 32) * sizeof(float); // ~257 KB
    attn_fused<<<BATCH * NHEAD * (SEQ / 32), 256, smem, stream>>>(
        Qh, Kh, Vt, mask, attn, Ob);

    // output projection + bias + residual, then LayerNorm in place
    fc_residual<<<512, 256, 0, stream>>>(Ob, Wft, b_fc, q, y);
    layernorm_k<<<ROWS, 256, 0, stream>>>(y, gamma, beta);
}
